// MemoryEdgeDecoder_84009560310089
// MI455X (gfx1250) — matrix operand order repaired and checked by reading, compiled, not run on hardware
//
#include <hip/hip_runtime.h>
#include <cstdint>
#include <cstddef>

// ---------------------------------------------------------------------------
// MemoryEdgeDecoder for MI455X (gfx1250).
// bf16-convert -> WMMA GEMM1(relu) -> WMMA GEMM2 -> gate epilogue ->
// 10x [WMMA GEMM3(relu) -> attn+weighted-kh reduce] -> changes/agg -> blend.
// Second einsum algebraically collapsed (sum_s attn == 1); all heavy math is
// v_wmma_f32_16x16x32_bf16. GEMM2 operands padded to N=2176 so every GEMM is
// an exact tile multiple (no bounds checks anywhere in the hot loop).
// Global->LDS staging uses GLOBAL_LOAD_ASYNC_TO_LDS_B128 (ASYNCcnt) with a
// double-buffered pipeline when the builtin exists; else register staging.
// ---------------------------------------------------------------------------

typedef int int4v __attribute__((vector_size(16)));   // matches builtin proto

#if defined(__has_builtin)
#if __has_builtin(__builtin_amdgcn_global_load_async_to_lds_b128)
#define HAVE_ASYNC_COPY 1
#endif
#endif
#ifndef HAVE_ASYNC_COPY
#define HAVE_ASYNC_COPY 0
#endif

#if HAVE_ASYNC_COPY
#if __has_builtin(__builtin_amdgcn_s_wait_asynccnt)
#define WAIT_ASYNC(n) __builtin_amdgcn_s_wait_asynccnt(n)
#else
#define WAIT_ASYNC(n) asm volatile("s_wait_asynccnt %0" ::"i"(n) : "memory")
#endif
#define ASYNC_CP_B128(gsrc, ldst)                                             \
  __builtin_amdgcn_global_load_async_to_lds_b128(                             \
      (__attribute__((address_space(1))) int4v*)(gsrc),                       \
      (__attribute__((address_space(3))) int4v*)(ldst), 0, 0)
#endif

typedef __attribute__((ext_vector_type(8)))  float          floatx8;
typedef __attribute__((ext_vector_type(16))) __bf16         bf16x16;
typedef __attribute__((ext_vector_type(8)))  unsigned short ushortx8;

union Frag {
  ushortx8 h[2];
  bf16x16  f;
};

__device__ __forceinline__ unsigned short f32_to_bf16(float x) {
  unsigned int u = __float_as_uint(x);
  u += 0x7fffu + ((u >> 16) & 1u);           // round-to-nearest-even
  return (unsigned short)(u >> 16);
}
__device__ __forceinline__ float bf16_to_f32(unsigned short h) {
  return __uint_as_float(((unsigned int)h) << 16);
}
__device__ __forceinline__ float sigmoidf_fast(float x) {
  return 1.0f / (1.0f + __expf(-x));
}

// ------------------------------ problem sizes ------------------------------
#define BB     32
#define SS     512
#define EE     512
#define MM     (BB * SS)        // 16384
#define INX    1024             // 2*E
#define HID    2048
#define EWM    32
#define OUTW   2080             // 4*E + EWM
#define OUTP   2176             // OUTW padded to 17*128 (tile multiple)
#define KK     10
#define RATE_F 0.005f

// ------------------------------ GEMM tiling --------------------------------
#define BM 128
#define BN 128
#define BK 32

// D = A(MxK,bf16) * B(KxN,bf16) + bias; optionally relu + bf16 store.
// Requires M%128==0, N%128==0, Kd%32==0 (guaranteed by padded buffers).
template <bool RELU_BF16>
__global__ __launch_bounds__(256) void gemm_bf16_wmma(
    const unsigned short* __restrict__ A,
    const unsigned short* __restrict__ Bw,
    const float* __restrict__ bias,
    void* __restrict__ Cv,
    int M, int N, int Kd)
{
#if HAVE_ASYNC_COPY
  __shared__ unsigned short As[2][BM][BK + 8];   // +8: stagger LDS banks
  __shared__ unsigned short Bs[2][BK][BN + 8];
#else
  __shared__ unsigned short As[1][BM][BK + 8];
  __shared__ unsigned short Bs[1][BK][BN + 8];
#endif

  const int tid  = threadIdx.x;
  const int lane = tid & 31;
  const int wid  = tid >> 5;
  const int wm   = (wid >> 2) << 6;           // wave M offset: 0 / 64
  const int wn   = (wid & 3) << 5;            // wave N offset: 0/32/64/96
  const int bm0  = blockIdx.y * BM;
  const int bn0  = blockIdx.x * BN;

  floatx8 acc[4][2] = {};

  // ISA A-fragment order per lane-half: K octets {0,16..}|{8,24..} ->
  // store octet o at slot kperm[o] so each half-lane reads 16B contiguous.
  const int kperm[4] = {0, 2, 1, 3};

  // per-thread staging assignment (2 A octets + 2 B octets per thread)
  const int ar0 = tid >> 2,         ak0 = tid & 3;           // A octet 0
  const int ar1 = (tid + 256) >> 2, ak1 = (tid + 256) & 3;   // A octet 1
  const int br0 = tid >> 4,         bc0 = tid & 15;          // B octet 0
  const int br1 = (tid + 256) >> 4, bc1 = (tid + 256) & 15;

  const int nk = Kd / BK;

#if HAVE_ASYNC_COPY
  // ---------------- async double-buffered pipeline ----------------
  {
    // issue tile 0 into buffer 0
    ASYNC_CP_B128(A + (size_t)(bm0 + ar0) * Kd + (ak0 << 3), &As[0][ar0][kperm[ak0] << 3]);
    ASYNC_CP_B128(A + (size_t)(bm0 + ar1) * Kd + (ak1 << 3), &As[0][ar1][kperm[ak1] << 3]);
    ASYNC_CP_B128(Bw + (size_t)br0 * N + bn0 + (bc0 << 3),   &Bs[0][br0][bc0 << 3]);
    ASYNC_CP_B128(Bw + (size_t)br1 * N + bn0 + (bc1 << 3),   &Bs[0][br1][bc1 << 3]);
  }
  for (int it = 0; it < nk; ++it) {
    const int cur = it & 1;
    const int k0  = it * BK;
    const bool has_next = (it + 1) < nk;
    if (has_next) {
      const int kn = k0 + BK;
      ASYNC_CP_B128(A + (size_t)(bm0 + ar0) * Kd + kn + (ak0 << 3), &As[cur ^ 1][ar0][kperm[ak0] << 3]);
      ASYNC_CP_B128(A + (size_t)(bm0 + ar1) * Kd + kn + (ak1 << 3), &As[cur ^ 1][ar1][kperm[ak1] << 3]);
      ASYNC_CP_B128(Bw + (size_t)(kn + br0) * N + bn0 + (bc0 << 3), &Bs[cur ^ 1][br0][bc0 << 3]);
      ASYNC_CP_B128(Bw + (size_t)(kn + br1) * N + bn0 + (bc1 << 3), &Bs[cur ^ 1][br1][bc1 << 3]);
      WAIT_ASYNC(4);            // own 4 ops for tile `it` complete (in-order)
    } else {
      WAIT_ASYNC(0);
    }
    __syncthreads();            // all waves' tile `it` writes visible

    Frag a[4], b[2];
#pragma unroll
    for (int i = 0; i < 4; ++i) {
      const unsigned short* p = &As[cur][wm + (i << 4) + (lane & 15)][(lane >> 4) << 4];
      a[i].h[0] = *(const ushortx8*)p;
      a[i].h[1] = *(const ushortx8*)(p + 8);
    }
#pragma unroll
    for (int j = 0; j < 2; ++j) {
      const int bn = wn + (j << 4) + (lane & 15);   // lane%16 == N col, K in table order
      const int kb = (lane >> 4) << 3;
#pragma unroll
      for (int s = 0; s < 8; ++s) {
        b[j].h[0][s] = Bs[cur][kb + s][bn];
        b[j].h[1][s] = Bs[cur][16 + kb + s][bn];
      }
    }
#pragma unroll
    for (int i = 0; i < 4; ++i)
#pragma unroll
      for (int j = 0; j < 2; ++j)
        acc[i][j] = __builtin_amdgcn_wmma_f32_16x16x32_bf16(
            false, a[i].f, false, b[j].f, (short)0, acc[i][j], false, false);
    __syncthreads();            // reads done before next overwrite
  }
#else
  // ---------------- register-staged software pipeline ----------------
  uint4 ra0, ra1, rb0, rb1;
  {
    ra0 = *(const uint4*)(A + (size_t)(bm0 + ar0) * Kd + (ak0 << 3));
    ra1 = *(const uint4*)(A + (size_t)(bm0 + ar1) * Kd + (ak1 << 3));
    rb0 = *(const uint4*)(Bw + (size_t)br0 * N + bn0 + (bc0 << 3));
    rb1 = *(const uint4*)(Bw + (size_t)br1 * N + bn0 + (bc1 << 3));
  }
  for (int it = 0; it < nk; ++it) {
    const int k0 = it * BK;
    *(uint4*)&As[0][ar0][kperm[ak0] << 3] = ra0;
    *(uint4*)&As[0][ar1][kperm[ak1] << 3] = ra1;
    *(uint4*)&Bs[0][br0][bc0 << 3] = rb0;
    *(uint4*)&Bs[0][br1][bc1 << 3] = rb1;
    __syncthreads();
    if (it + 1 < nk) {          // overlap next fetch with WMMA
      const int kn = k0 + BK;
      ra0 = *(const uint4*)(A + (size_t)(bm0 + ar0) * Kd + kn + (ak0 << 3));
      ra1 = *(const uint4*)(A + (size_t)(bm0 + ar1) * Kd + kn + (ak1 << 3));
      rb0 = *(const uint4*)(Bw + (size_t)(kn + br0) * N + bn0 + (bc0 << 3));
      rb1 = *(const uint4*)(Bw + (size_t)(kn + br1) * N + bn0 + (bc1 << 3));
    }
    Frag a[4], b[2];
#pragma unroll
    for (int i = 0; i < 4; ++i) {
      const unsigned short* p = &As[0][wm + (i << 4) + (lane & 15)][(lane >> 4) << 4];
      a[i].h[0] = *(const ushortx8*)p;
      a[i].h[1] = *(const ushortx8*)(p + 8);
    }
#pragma unroll
    for (int j = 0; j < 2; ++j) {
      const int bn = wn + (j << 4) + (lane & 15);
      const int kb = (lane >> 4) << 3;
#pragma unroll
      for (int s = 0; s < 8; ++s) {
        b[j].h[0][s] = Bs[0][kb + s][bn];
        b[j].h[1][s] = Bs[0][16 + kb + s][bn];
      }
    }
#pragma unroll
    for (int i = 0; i < 4; ++i)
#pragma unroll
      for (int j = 0; j < 2; ++j)
        acc[i][j] = __builtin_amdgcn_wmma_f32_16x16x32_bf16(
            false, a[i].f, false, b[j].f, (short)0, acc[i][j], false, false);
    __syncthreads();
  }
#endif

  // ---- epilogue: C layout -> VGPR e: row = e + (lane>=16 ? 8 : 0) ----
  const int cm = (lane >> 4) << 3;
  const int cn = lane & 15;
#pragma unroll
  for (int i = 0; i < 4; ++i) {
#pragma unroll
    for (int j = 0; j < 2; ++j) {
      int col = bn0 + wn + (j << 4) + cn;
      float bv = bias[col];
#pragma unroll
      for (int e = 0; e < 8; ++e) {
        int row = bm0 + wm + (i << 4) + cm + e;
        float v = acc[i][j][e] + bv;
        if constexpr (RELU_BF16) {
          v = v > 0.0f ? v : 0.0f;
          ((unsigned short*)Cv)[(size_t)row * N + col] = f32_to_bf16(v);
        } else {
          ((float*)Cv)[(size_t)row * N + col] = v;
        }
      }
    }
  }
}

// ---------------------------- elementwise kernels --------------------------
__global__ __launch_bounds__(256) void cvt_bf16_kernel(
    const float* __restrict__ src, unsigned short* __restrict__ dst, int n)
{
  int i = blockIdx.x * 256 + threadIdx.x;
  if (i < n) dst[i] = f32_to_bf16(src[i]);
}

// row-padded convert: dst[r][0:cols)=bf16(src), dst[r][cols:ldd)=0
__global__ __launch_bounds__(256) void cvt_pad_bf16_kernel(
    const float* __restrict__ src, unsigned short* __restrict__ dst,
    int cols, int ldd)
{
  int r = blockIdx.x;
  for (int c = threadIdx.x; c < ldd; c += 256)
    dst[(size_t)r * ldd + c] =
        (c < cols) ? f32_to_bf16(src[(size_t)r * cols + c]) : (unsigned short)0;
}

__global__ __launch_bounds__(256) void pad_bias_kernel(
    const float* __restrict__ src, float* __restrict__ dst, int n, int np)
{
  int i = blockIdx.x * 256 + threadIdx.x;
  if (i < np) dst[i] = (i < n) ? src[i] : 0.0f;
}

__global__ __launch_bounds__(256) void pack_x_kernel(
    const float* __restrict__ l, const float* __restrict__ r,
    unsigned short* __restrict__ xbf)
{
  int idx = blockIdx.x * 256 + threadIdx.x;        // over 16384*1024
  int m = idx >> 10, j = idx & 1023;
  float v = (j < EE) ? l[(size_t)m * EE + j] : r[(size_t)m * EE + (j - EE)];
  xbf[idx] = f32_to_bf16(v);
}

// out(16384 x OUTP) -> decoded (d_out), dbl f32, dbl bf16 (sigmoid gate)
__global__ __launch_bounds__(256) void gate_epilogue_kernel(
    const float* __restrict__ out, const float* __restrict__ l,
    const float* __restrict__ r, float* __restrict__ decoded,
    float* __restrict__ dblf, unsigned short* __restrict__ dblbf)
{
  int m = blockIdx.x;
  const float* row = out + (size_t)m * OUTP;
  for (int j = threadIdx.x; j < EWM + INX; j += 256) {
    if (j < EWM) {
      decoded[(size_t)m * EWM + j] = row[j];
    } else {
      int jj = j - EWM;
      float d    = row[EWM + jj];
      float g    = sigmoidf_fast(row[EWM + INX + jj]);
      float prev = (jj < EE) ? l[(size_t)m * EE + jj] : r[(size_t)m * EE + (jj - EE)];
      float v    = d * g + prev * (1.0f - g);
      dblf[(size_t)m * INX + jj]  = v;
      dblbf[(size_t)m * INX + jj] = f32_to_bf16(v);
    }
  }
}

// per (b): attn logits over S from kh . w2[:,last], normalize, weighted kh sum
__global__ __launch_bounds__(256) void attn_wkh_kernel(
    const unsigned short* __restrict__ kh,      // 16384 x 1024 bf16, this k
    const float* __restrict__ ker_w2,           // (K, IN, IN+1)
    const float* __restrict__ ker_b2,           // (K, IN+1)
    float* __restrict__ wkh,                    // (B, K, IN)
    int k)
{
  __shared__ float attn[SS];
  __shared__ float w2c[INX];
  __shared__ float red[9];
  const int b = blockIdx.x, tid = threadIdx.x, lane = tid & 31, wv = tid >> 5;

  for (int h = tid; h < INX; h += 256)
    w2c[h] = ker_w2[((size_t)k * INX + h) * (INX + 1) + INX];
  __syncthreads();

  const float blast = ker_b2[(size_t)k * (INX + 1) + INX];
  for (int s = wv; s < SS; s += 8) {
    const unsigned short* row = kh + ((size_t)(b * SS + s)) * INX;
    float acc = 0.0f;
    for (int h = lane; h < INX; h += 32) acc += bf16_to_f32(row[h]) * w2c[h];
#pragma unroll
    for (int off = 16; off > 0; off >>= 1) acc += __shfl_xor(acc, off, 32);
    if (lane == 0) attn[s] = sigmoidf_fast(acc + blast);
  }
  __syncthreads();

  float p = attn[tid] + attn[tid + 256];
#pragma unroll
  for (int off = 16; off > 0; off >>= 1) p += __shfl_xor(p, off, 32);
  if (lane == 0) red[wv] = p;
  __syncthreads();
  if (tid == 0) {
    float t = 0.0f;
    for (int i = 0; i < 8; ++i) t += red[i];
    red[8] = 1.0f / t;
  }
  __syncthreads();
  const float inv = red[8];

  for (int h = tid; h < INX; h += 256) {
    float acc = 0.0f;
    for (int s = 0; s < SS; ++s)
      acc += attn[s] * bf16_to_f32(kh[((size_t)(b * SS + s)) * INX + h]);
    wkh[((size_t)b * KK + k) * INX + h] = acc * inv;
  }
}

// agg[b,o] = mean_k ( b2[k,o] + sum_h wkh[b,k,h] * w2[k,h,o] )
__global__ __launch_bounds__(256) void changes_agg_kernel(
    const float* __restrict__ wkh, const float* __restrict__ ker_w2,
    const float* __restrict__ ker_b2, float* __restrict__ agg)
{
  __shared__ float w[KK * INX];                  // 40 KB, fine on 320KB WGP LDS
  const int b = blockIdx.x, tid = threadIdx.x;
  for (int i = tid; i < KK * INX; i += 256) w[i] = wkh[(size_t)b * KK * INX + i];
  __syncthreads();
  for (int o = tid; o < INX; o += 256) {
    float acc = 0.0f;
    for (int k = 0; k < KK; ++k) {
      float c = ker_b2[(size_t)k * (INX + 1) + o];
      const float* w2k = ker_w2 + (size_t)k * INX * (INX + 1) + o;
      const float* wk  = w + k * INX;
      for (int h = 0; h < INX; ++h) c += wk[h] * w2k[(size_t)h * (INX + 1)];
      acc += c;
    }
    agg[(size_t)b * INX + o] = acc * (1.0f / KK);
  }
}

__global__ __launch_bounds__(256) void final_blend_kernel(
    const float* __restrict__ dblf, const float* __restrict__ agg,
    float* __restrict__ newl, float* __restrict__ newr)
{
  int idx = blockIdx.x * 256 + threadIdx.x;      // over 16384*1024
  int m = idx >> 10, j = idx & 1023, b = m >> 9;
  float v = dblf[idx] * (1.0f - RATE_F) + agg[(size_t)b * INX + j] * RATE_F;
  if (j < EE) newl[(size_t)m * EE + j] = v;
  else        newr[(size_t)m * EE + (j - EE)] = v;
}

// ------------------------------- launcher ----------------------------------
extern "C" void kernel_launch(void* const* d_in, const int* in_sizes, int n_in,
                              void* d_out, int out_size, void* d_ws, size_t ws_size,
                              hipStream_t stream)
{
  const float* embl   = (const float*)d_in[0];
  const float* embr   = (const float*)d_in[1];
  const float* nn_w1  = (const float*)d_in[2];
  const float* nn_b1  = (const float*)d_in[3];
  const float* nn_w2  = (const float*)d_in[4];
  const float* nn_b2  = (const float*)d_in[5];
  const float* ker_w1 = (const float*)d_in[6];
  const float* ker_b1 = (const float*)d_in[7];
  const float* ker_w2 = (const float*)d_in[8];
  const float* ker_b2 = (const float*)d_in[9];

  char* ws = (char*)d_ws;
  constexpr size_t OFF_XBF   = 0;                              // MMxINX bf16
  constexpr size_t OFF_W1BF  = OFF_XBF   + (size_t)MM  * INX  * 2;
  constexpr size_t OFF_W2BF  = OFF_W1BF  + (size_t)INX * HID  * 2;   // HIDxOUTP
  constexpr size_t OFF_B2P   = OFF_W2BF  + (size_t)HID * OUTP * 2;   // OUTP f32
  constexpr size_t OFF_KW1BF = OFF_B2P   + (size_t)OUTP * 4;
  constexpr size_t OFF_HBF   = OFF_KW1BF + (size_t)KK  * INX * INX * 2;
  constexpr size_t OFF_OUTF  = OFF_HBF   + (size_t)MM  * HID  * 2;   // MMxOUTP
  constexpr size_t OFF_DBLF  = OFF_OUTF  + (size_t)MM  * OUTP * 4;
  constexpr size_t OFF_DBLBF = OFF_DBLF  + (size_t)MM  * INX  * 4;
  constexpr size_t OFF_KH    = OFF_DBLBF + (size_t)MM  * INX  * 2;
  constexpr size_t OFF_WKH   = OFF_KH    + (size_t)MM  * INX  * 2;
  constexpr size_t OFF_AGG   = OFF_WKH   + (size_t)BB  * KK * INX * 4;

  unsigned short* XBF   = (unsigned short*)(ws + OFF_XBF);
  unsigned short* W1BF  = (unsigned short*)(ws + OFF_W1BF);
  unsigned short* W2BF  = (unsigned short*)(ws + OFF_W2BF);
  float*          B2P   = (float*)(ws + OFF_B2P);
  unsigned short* KW1BF = (unsigned short*)(ws + OFF_KW1BF);
  unsigned short* HBF   = (unsigned short*)(ws + OFF_HBF);
  float*          OUTF  = (float*)(ws + OFF_OUTF);
  float*          DBLF  = (float*)(ws + OFF_DBLF);
  unsigned short* DBLBF = (unsigned short*)(ws + OFF_DBLBF);
  unsigned short* KH    = (unsigned short*)(ws + OFF_KH);
  float*          WKH   = (float*)(ws + OFF_WKH);
  float*          AGG   = (float*)(ws + OFF_AGG);

  float* decoded = (float*)d_out;                              // MM*EWM
  float* newl    = decoded + (size_t)MM * EWM;                 // MM*EE
  float* newr    = newl    + (size_t)MM * EE;                  // MM*EE

  // 1) weight / activation conversion to bf16 (W2 & bias padded to OUTP)
  {
    int n1 = INX * HID;
    cvt_bf16_kernel<<<(n1 + 255) / 256, 256, 0, stream>>>(nn_w1, W1BF, n1);
    cvt_pad_bf16_kernel<<<HID, 256, 0, stream>>>(nn_w2, W2BF, OUTW, OUTP);
    pad_bias_kernel<<<(OUTP + 255) / 256, 256, 0, stream>>>(nn_b2, B2P, OUTW, OUTP);
    int n3 = KK * INX * INX;
    cvt_bf16_kernel<<<(n3 + 255) / 256, 256, 0, stream>>>(ker_w1, KW1BF, n3);
    pack_x_kernel<<<(MM * INX) / 256, 256, 0, stream>>>(embl, embr, XBF);
  }

  // 2) GEMM1: H = relu(X*W1 + b1), bf16 out
  gemm_bf16_wmma<true><<<dim3(HID / BN, MM / BM), 256, 0, stream>>>(
      XBF, W1BF, nn_b1, HBF, MM, HID, INX);

  // 3) GEMM2: OUT = H*W2p + b2p, f32 out, padded N
  gemm_bf16_wmma<false><<<dim3(OUTP / BN, MM / BM), 256, 0, stream>>>(
      HBF, W2BF, B2P, OUTF, MM, OUTP, HID);

  // 4) decoded + sigmoid-gated dbl
  gate_epilogue_kernel<<<MM, 256, 0, stream>>>(OUTF, embl, embr, decoded, DBLF, DBLBF);

  // 5) per k: kh = relu(dbl*W1k + b1k); attn + weighted-kh reduce
  for (int k = 0; k < KK; ++k) {
    gemm_bf16_wmma<true><<<dim3(INX / BN, MM / BM), 256, 0, stream>>>(
        DBLBF, KW1BF + (size_t)k * INX * INX, ker_b1 + (size_t)k * INX,
        KH, MM, INX, INX);
    attn_wkh_kernel<<<BB, 256, 0, stream>>>(KH, ker_w2, ker_b2, WKH, k);
  }

  // 6) changes -> agg (mean over k)   7) final blend into new_l / new_r
  changes_agg_kernel<<<BB, 256, 0, stream>>>(WKH, ker_w2, ker_b2, AGG);
  final_blend_kernel<<<(MM * INX) / 256, 256, 0, stream>>>(DBLF, AGG, newl, newr);

  (void)in_sizes; (void)n_in; (void)out_size; (void)ws_size;
}
